// BlockVanilla_31447750541459
// MI455X (gfx1250) — matrix operand order repaired and checked by reading, not compiled, not run on hardware
//
#include <hip/hip_runtime.h>
#include <hip/hip_bf16.h>

// ---------------------------------------------------------------------------
// Transformer block (pre-LN, MHA + GELU FFN) for gfx1250 (MI455X, CDNA5).
// All matmuls: v_wmma_f32_16x16x32_bf16 (bf16 in, fp32 accumulate).
// - Weights pre-transposed once to [N][K] bf16; GEMM 128x128 tile, K-step 64,
//   8 waves, 16 WMMA per wave per stage, register double-buffered staging.
// - Attention computes TRANSPOSED products (scores^T = K Q^T, O^T = V^T P^T)
//   so softmax statistics are per-lane (1 shuffle instead of 4 per row), P^T
//   is built in registers (8 half-swap shuffles), zero LDS, zero barriers.
// Workspace requirement: ~216 MB.
// ---------------------------------------------------------------------------

typedef __bf16 bf16;
typedef __attribute__((ext_vector_type(16))) __bf16 v16bf;
typedef __attribute__((ext_vector_type(8)))  __bf16 v8bf;
typedef __attribute__((ext_vector_type(8)))  float  v8f;

#define WMMA_BF16(a, b, c) \
  __builtin_amdgcn_wmma_f32_16x16x32_bf16(false, (a), false, (b), (short)0, (c), false, false)

// ---------------------------------------------------------------------------
// fp32 [K][N] -> bf16 transposed [N][K], 32x32 LDS-tiled
// ---------------------------------------------------------------------------
__global__ __launch_bounds__(256) void k_cvt_T(const float* __restrict__ src,
                                               bf16* __restrict__ dst, int K, int N) {
  __shared__ float t[32][33];
  int kt = blockIdx.y * 32, nt = blockIdx.x * 32;
  int lx = threadIdx.x & 31, ly = threadIdx.x >> 5;
#pragma unroll
  for (int r = ly; r < 32; r += 8) t[r][lx] = src[(size_t)(kt + r) * N + nt + lx];
  __syncthreads();
#pragma unroll
  for (int r = ly; r < 32; r += 8) dst[(size_t)(nt + r) * K + kt + lx] = (bf16)t[lx][r];
}

// ---------------------------------------------------------------------------
// LayerNorm over last dim D; fp32 in, bf16 out. One block per row.
// ---------------------------------------------------------------------------
__global__ __launch_bounds__(256) void k_layernorm(const float* __restrict__ x,
                                                   const float* __restrict__ g,
                                                   const float* __restrict__ bta,
                                                   bf16* __restrict__ out, int D) {
  __shared__ float red[256];
  int row = blockIdx.x;
  const float* xr = x + (size_t)row * D;
  float s = 0.f, s2 = 0.f;
  for (int i = threadIdx.x; i < D; i += 256) { float v = xr[i]; s += v; s2 += v * v; }
  red[threadIdx.x] = s; __syncthreads();
  for (int o = 128; o > 0; o >>= 1) { if (threadIdx.x < o) red[threadIdx.x] += red[threadIdx.x + o]; __syncthreads(); }
  float mean = red[0] / D;
  __syncthreads();
  red[threadIdx.x] = s2; __syncthreads();
  for (int o = 128; o > 0; o >>= 1) { if (threadIdx.x < o) red[threadIdx.x] += red[threadIdx.x + o]; __syncthreads(); }
  float var = red[0] / D - mean * mean;
  float inv = rsqrtf(var + 1e-6f);
  for (int i = threadIdx.x; i < D; i += 256)
    out[(size_t)row * D + i] = (bf16)((xr[i] - mean) * inv * g[i] + bta[i]);
}

// ---------------------------------------------------------------------------
// Tiled bf16 GEMM: C[M,N] = act( A[M,K] @ Bt[N,K]^T + bias [+ resid] )
// Block tile 128x128, K-step 64, 8 waves (4m x 2n), 16 WMMAs per wave/stage.
// A tile stored in natural k order so an A-fragment is ONE contiguous v16bf read.
// MODE 0: bf16 out;  1: fp32 out = resid + v;  2: GELU -> bf16;
// MODE 3: bf16 out transposed to [b, head, dim, s]   (for attention V)
// ---------------------------------------------------------------------------
#define BM 128
#define BN 128
#define BK 64
#define LSTR 80   // 160 B row stride: multiple of 32 B for aligned v16bf reads

template <int MODE>
__global__ __launch_bounds__(256) void k_gemm(const bf16* __restrict__ A,
                                              const bf16* __restrict__ Bt,
                                              const float* __restrict__ bias,
                                              const float* __restrict__ resid,
                                              void* __restrict__ Cout,
                                              int M, int N, int K, int Sdim) {
  __shared__ alignas(64) bf16 As[BM][LSTR];   // 20 KB
  __shared__ alignas(64) bf16 Bs[BN][LSTR];   // 20 KB

  const int tid  = threadIdx.x;
  const int lane = tid & 31;
  const int wid  = tid >> 5;
  const int wm   = wid & 3;
  const int wn   = wid >> 2;
  const int m0   = blockIdx.y * BM;
  const int n0   = blockIdx.x * BN;
  const int ml   = lane & 15;
  const int half = lane >> 4;

  const int srow = tid >> 1;
  const int sh   = tid & 1;
  const bf16* gA = A  + (size_t)(m0 + srow) * K + sh * 32;
  const bf16* gB = Bt + (size_t)(n0 + srow) * K + sh * 32;

  v8bf a_st[4], b_st[4];
  auto stage = [&](int k0) {
#pragma unroll
    for (int c = 0; c < 4; c++) a_st[c] = *(const v8bf*)(gA + k0 + c * 8);
#pragma unroll
    for (int c = 0; c < 4; c++) b_st[c] = *(const v8bf*)(gB + k0 + c * 8);
  };
  auto commit = [&]() {
    bf16* dA = &As[srow][sh * 32];
    *(v8bf*)(dA + 0)  = a_st[0];   // k0-7
    *(v8bf*)(dA + 8)  = a_st[1];   // k8-15
    *(v8bf*)(dA + 16) = a_st[2];   // k16-23
    *(v8bf*)(dA + 24) = a_st[3];   // k24-31
    bf16* dB = &Bs[srow][sh * 32];
#pragma unroll
    for (int c = 0; c < 4; c++) *(v8bf*)(dB + c * 8) = b_st[c];
  };

  v8f acc[2][4] = {};

  stage(0);
  commit();
  __syncthreads();

  for (int k0 = 0; k0 < K; k0 += BK) {
    const bool more = (k0 + BK) < K;
    if (more) stage(k0 + BK);               // global loads in flight over WMMAs

#pragma unroll
    for (int ks = 0; ks < 2; ks++) {
      v16bf af[2], bfv[4];
#pragma unroll
      for (int i = 0; i < 2; i++)
        af[i] = *(const v16bf*)(&As[wm * 32 + i * 16 + ml][ks * 32 + half * 16]);
#pragma unroll
      for (int j = 0; j < 4; j++)
        bfv[j] = *(const v16bf*)(&Bs[wn * 64 + j * 16 + ml][ks * 32 + half * 16]);
#pragma unroll
      for (int i = 0; i < 2; i++)
#pragma unroll
        for (int j = 0; j < 4; j++)
          acc[i][j] = WMMA_BF16(af[i], bfv[j], acc[i][j]);
    }

    if (more) {
      __syncthreads();
      commit();
      __syncthreads();
    }
  }

#pragma unroll
  for (int i = 0; i < 2; i++)
#pragma unroll
    for (int j = 0; j < 4; j++)
#pragma unroll
      for (int r = 0; r < 8; r++) {
        int row = m0 + wm * 32 + i * 16 + r + 8 * half;
        int col = n0 + wn * 64 + j * 16 + ml;
        float v = acc[i][j][r] + bias[col];
        if (MODE == 0) {
          ((bf16*)Cout)[(size_t)row * N + col] = (bf16)v;
        } else if (MODE == 1) {
          size_t idx = (size_t)row * N + col;
          ((float*)Cout)[idx] = resid[idx] + v;
        } else if (MODE == 2) {
          float ge = 0.5f * v * (1.0f + erff(v * 0.70710678118654752f));
          ((bf16*)Cout)[(size_t)row * N + col] = (bf16)ge;
        } else {                  // MODE 3: [b, head, dim, s] for attention V
          int gb = row / Sdim, ss = row % Sdim;
          int hh = col >> 6, dd = col & 63;
          size_t idx = ((((size_t)gb * (N >> 6) + hh) * 64 + dd) * Sdim) + ss;
          ((bf16*)Cout)[idx] = (bf16)v;
        }
      }
}

// ---------------------------------------------------------------------------
// Flash attention, transposed formulation. One wave per 32-query tile per
// (b,h); hd=64, scale=1/8.
//   scores^T = K_tile(16k x 64d) @ Q^T(64d x 16q): K rows are A-fragments,
//     Q rows are B-fragments, both straight from global (aligned v16bf/v8bf).
//   C layout of scores^T: reg r -> key r+8*half, lane -> query. So softmax
//     max/sum are in-lane over 16 regs + ONE shfl_xor(16) to merge halves.
//   P^T built in registers with 8 half-swap shuffles; O^T = V^T P^T with
//     Vt[b,h,dim,s] rows as A-fragments. No LDS, no barriers.
// ---------------------------------------------------------------------------
__global__ __launch_bounds__(32) void k_attn(const bf16* __restrict__ Q,
                                             const bf16* __restrict__ Km,
                                             const bf16* __restrict__ Vt,
                                             bf16* __restrict__ O,
                                             int S, int D, int H) {
  const int q0   = blockIdx.x * 32;
  const int h    = blockIdx.y;
  const int b    = blockIdx.z;
  const int lane = threadIdx.x;
  const int nl   = lane & 15;
  const int half = lane >> 4;
  const int HD   = 64;
  const float scale = 0.125f;  // 1/sqrt(64)

  // Q B-fragments: [qtile][ks], lane = query, contiguous dims
  v16bf qbf[2][2];
#pragma unroll
  for (int qt = 0; qt < 2; qt++)
#pragma unroll
    for (int ks = 0; ks < 2; ks++) {
      const bf16* qp = Q + ((size_t)b * S + q0 + qt * 16 + nl) * D + h * HD + ks * 32 + 16 * half;
      qbf[qt][ks] = *(const v16bf*)qp;
    }

  // running pointers (avoid per-iter 64-bit muls)
  const bf16* kptr[2];
#pragma unroll
  for (int t = 0; t < 2; t++)
    kptr[t] = Km + ((size_t)b * S + t * 16 + nl) * D + h * HD;
  const bf16* vptr[4];
#pragma unroll
  for (int dt = 0; dt < 4; dt++)
    vptr[dt] = Vt + (((size_t)b * H + h) * HD + dt * 16 + nl) * S;
  const size_t kStep = (size_t)32 * D;

  v8f   o[2][4] = {};
  float rm[2] = {-1e30f, -1e30f}, rl[2] = {0.f, 0.f};

  for (int kt = 0; kt < S; kt += 32) {
    if (kt + 32 < S) {   // gfx1250 global_prefetch_b8 for next K/V tiles
      __builtin_prefetch(kptr[0] + kStep, 0, 1);
      __builtin_prefetch(kptr[1] + kStep, 0, 1);
      __builtin_prefetch(vptr[half * 2] + 32, 0, 1);
    }

    // K A-fragments [keytile][ks]; chunks 16*half / 16*half+8
    v16bf ka[2][2];
#pragma unroll
    for (int t = 0; t < 2; t++)
#pragma unroll
      for (int ks = 0; ks < 2; ks++) {
        const bf16* kp = kptr[t] + ks * 32;
        v8bf lo = *(const v8bf*)(kp + 16 * half);
        v8bf hi = *(const v8bf*)(kp + 16 * half + 8);
#pragma unroll
        for (int e = 0; e < 8; e++) { ka[t][ks][e] = lo[e]; ka[t][ks][e + 8] = hi[e]; }
      }
    // V^T A-fragments [dimtile]; keys contiguous along Vt rows
    v16bf va[4];
#pragma unroll
    for (int dt = 0; dt < 4; dt++) {
      const bf16* vp = vptr[dt] + kt;
      v8bf lo = *(const v8bf*)(vp + 16 * half);
      v8bf hi = *(const v8bf*)(vp + 16 * half + 8);
#pragma unroll
      for (int e = 0; e < 8; e++) { va[dt][e] = lo[e]; va[dt][e + 8] = hi[e]; }
    }

#pragma unroll
    for (int qt = 0; qt < 2; qt++) {
      // scores^T: 4 WMMAs
      v8f sct[2] = {};
#pragma unroll
      for (int t = 0; t < 2; t++)
#pragma unroll
        for (int ks = 0; ks < 2; ks++)
          sct[t] = WMMA_BF16(ka[t][ks], qbf[qt][ks], sct[t]);

      // per-lane online softmax over this lane's 16 keys + partner half
      float sv[16];
      float smax = -1e30f;
#pragma unroll
      for (int t = 0; t < 2; t++)
#pragma unroll
        for (int r = 0; r < 8; r++) {
          float s = sct[t][r] * scale;
          sv[t * 8 + r] = s;
          smax = fmaxf(smax, s);
        }
      smax = fmaxf(smax, __shfl_xor(smax, 16, 32));
      float mn = fmaxf(rm[qt], smax);
      float psum = 0.f;
#pragma unroll
      for (int i = 0; i < 16; i++) { sv[i] = __expf(sv[i] - mn); psum += sv[i]; }
      psum += __shfl_xor(psum, 16, 32);
      float f = __expf(rm[qt] - mn);
      rl[qt] = rl[qt] * f + psum;
      rm[qt] = mn;
#pragma unroll
      for (int dt = 0; dt < 4; dt++)
#pragma unroll
        for (int r = 0; r < 8; r++) o[qt][dt][r] *= f;

      // build P^T B-fragment: lane=query, k=key=e+16*half.
      // half0 lane holds keys {0-7,16-23}, half1 holds {8-15,24-31};
      // one xor-16 shuffle per reg swaps the off-half chunk into place.
      v16bf pbf;
#pragma unroll
      for (int r = 0; r < 8; r++) {
        float send = half ? sv[r] : sv[8 + r];
        float recv = __shfl_xor(send, 16, 32);
        float e_lo = half ? recv : sv[r];        // e = r     (key r + 16*half)
        float e_hi = half ? sv[8 + r] : recv;    // e = 8 + r (key 8+r+16*half)
        pbf[r]     = (bf16)e_lo;
        pbf[8 + r] = (bf16)e_hi;
      }

      // O^T += V^T @ P^T : 4 WMMAs
#pragma unroll
      for (int dt = 0; dt < 4; dt++)
        o[qt][dt] = WMMA_BF16(va[dt], pbf, o[qt][dt]);
    }

#pragma unroll
    for (int t = 0; t < 2; t++) kptr[t] += kStep;
  }

  // epilogue: lane owns query q0+qt*16+nl; frag dt reg r -> dim dt*16+r+8*half
  // -> 8 consecutive dims per (dt,half): one packed 16-B store each.
#pragma unroll
  for (int qt = 0; qt < 2; qt++) {
    float inv = 1.0f / rl[qt];
    bf16* op = O + ((size_t)b * S + q0 + qt * 16 + nl) * D + h * HD + 8 * half;
#pragma unroll
    for (int dt = 0; dt < 4; dt++) {
      v8bf pack;
#pragma unroll
      for (int r = 0; r < 8; r++) pack[r] = (bf16)(o[qt][dt][r] * inv);
      *(v8bf*)(op + dt * 16) = pack;
    }
  }
}

// ---------------------------------------------------------------------------
// Host launcher
// ---------------------------------------------------------------------------
extern "C" void kernel_launch(void* const* d_in, const int* in_sizes, int n_in,
                              void* d_out, int out_size, void* d_ws, size_t ws_size,
                              hipStream_t stream) {
  const int B = 4, S = 2048, D = 1024, H = 16, FF = 4096;
  const int M = B * S;

  const float* x   = (const float*)d_in[0];
  const float* Wq  = (const float*)d_in[1];
  const float* bq  = (const float*)d_in[2];
  const float* Wk  = (const float*)d_in[3];
  const float* bk  = (const float*)d_in[4];
  const float* Wv  = (const float*)d_in[5];
  const float* bv  = (const float*)d_in[6];
  const float* Wo  = (const float*)d_in[7];
  const float* bo  = (const float*)d_in[8];
  const float* g1  = (const float*)d_in[9];
  const float* b1  = (const float*)d_in[10];
  const float* W1  = (const float*)d_in[11];
  const float* bf1 = (const float*)d_in[12];
  const float* W2  = (const float*)d_in[13];
  const float* bf2 = (const float*)d_in[14];
  const float* g2  = (const float*)d_in[15];
  const float* b2  = (const float*)d_in[16];

  char* w = (char*)d_ws;
  size_t off = 0;
  auto take = [&](size_t bytes) { char* p = w + off; off += (bytes + 255) & ~(size_t)255; return p; };

  bf16* wq_t  = (bf16*)take((size_t)D * D * 2);    // all weights bf16, [N][K]
  bf16* wk_t  = (bf16*)take((size_t)D * D * 2);
  bf16* wv_t  = (bf16*)take((size_t)D * D * 2);
  bf16* wo_t  = (bf16*)take((size_t)D * D * 2);
  bf16* w1_t  = (bf16*)take((size_t)D * FF * 2);
  bf16* w2_t  = (bf16*)take((size_t)FF * D * 2);
  bf16* xn_b  = (bf16*)take((size_t)M * D * 2);
  bf16* q_b   = (bf16*)take((size_t)M * D * 2);
  bf16* k_b   = (bf16*)take((size_t)M * D * 2);
  bf16* vt_b  = (bf16*)take((size_t)M * D * 2);    // V transposed [b,h,dim,s]
  bf16* h_b   = (bf16*)take((size_t)M * D * 2);
  float* x1   = (float*)take((size_t)M * D * 4);
  bf16* xn2_b = (bf16*)take((size_t)M * D * 2);
  bf16* ff_b  = (bf16*)take((size_t)M * FF * 2);
  (void)ws_size; (void)in_sizes; (void)n_in; (void)out_size;

  // 1) weights -> bf16 transposed [N][K]
  k_cvt_T<<<dim3(D / 32, D / 32),  256, 0, stream>>>(Wq, wq_t, D, D);
  k_cvt_T<<<dim3(D / 32, D / 32),  256, 0, stream>>>(Wk, wk_t, D, D);
  k_cvt_T<<<dim3(D / 32, D / 32),  256, 0, stream>>>(Wv, wv_t, D, D);
  k_cvt_T<<<dim3(D / 32, D / 32),  256, 0, stream>>>(Wo, wo_t, D, D);
  k_cvt_T<<<dim3(FF / 32, D / 32), 256, 0, stream>>>(W1, w1_t, D, FF);
  k_cvt_T<<<dim3(D / 32, FF / 32), 256, 0, stream>>>(W2, w2_t, FF, D);

  // 2) LN1
  k_layernorm<<<M, 256, 0, stream>>>(x, g1, b1, xn_b, D);

  // 3) Q/K projections (bf16), V projection written pre-transposed
  dim3 gDD(D / BN, M / BM);
  k_gemm<0><<<gDD, 256, 0, stream>>>(xn_b, wq_t, bq, nullptr, (void*)q_b,  M, D, D, S);
  k_gemm<0><<<gDD, 256, 0, stream>>>(xn_b, wk_t, bk, nullptr, (void*)k_b,  M, D, D, S);
  k_gemm<3><<<gDD, 256, 0, stream>>>(xn_b, wv_t, bv, nullptr, (void*)vt_b, M, D, D, S);

  // 4) flash attention (transposed formulation)
  k_attn<<<dim3(S / 32, H, B), 32, 0, stream>>>(q_b, k_b, vt_b, h_b, S, D, H);

  // 5) output projection + residual 1 (fp32)
  k_gemm<1><<<gDD, 256, 0, stream>>>(h_b, wo_t, bo, x, (void*)x1, M, D, D, S);

  // 6) LN2
  k_layernorm<<<M, 256, 0, stream>>>(x1, g2, b2, xn2_b, D);

  // 7) FFN up + exact GELU (bf16 out)
  k_gemm<2><<<dim3(FF / BN, M / BM), 256, 0, stream>>>(xn2_b, w1_t, bf1, nullptr, (void*)ff_b, M, FF, D, S);

  // 8) FFN down + residual 2 -> d_out (fp32)
  k_gemm<1><<<gDD, 256, 0, stream>>>(ff_b, w2_t, bf2, x1, d_out, M, D, FF, S);
}
